// CausalSelfAttention_90391881711741
// MI455X (gfx1250) — matrix operand order repaired and checked by reading, not compiled, not run on hardware
//
#include <hip/hip_runtime.h>
#include <hip/hip_bf16.h>
#include <stdint.h>

typedef __attribute__((ext_vector_type(16))) __bf16 v16bf;
typedef __attribute__((ext_vector_type(8)))  __bf16 v8bf;
typedef __attribute__((ext_vector_type(8)))  float  v8f;
typedef __attribute__((ext_vector_type(4)))  float  v4f;

#define Bq   8
#define Tq   2048
#define Cq   1024
#define Hq   16
#define HDq  64

static __device__ __forceinline__ v16bf frag16(const __bf16* lo_p, const __bf16* hi_p) {
  v8bf lo = *(const v8bf*)lo_p;
  v8bf hi = *(const v8bf*)hi_p;
  return __builtin_shufflevector(lo, hi, 0,1,2,3,4,5,6,7,8,9,10,11,12,13,14,15);
}

static __device__ __forceinline__ v8f wmma_bf16(v16bf a, v16bf b, v8f c) {
  return __builtin_amdgcn_wmma_f32_16x16x32_bf16(false, a, false, b, (short)0, c, false, false);
}

// Async global->LDS 16-byte copy (per lane), tracked by ASYNCcnt (ISA 15.18.3 op 98).
static __device__ __forceinline__ void async_cp_b128(void* lds, const void* gsrc) {
  unsigned l = (unsigned)(uintptr_t)lds;   // low 32 bits of generic shared ptr = LDS offset
  asm volatile("global_load_async_to_lds_b128 %0, %1, off"
               :: "v"(l), "v"(gsrc) : "memory");
}
static __device__ __forceinline__ void wait_async0() {
  asm volatile("s_wait_asynccnt 0x0" ::: "memory");
}

// ---------------------------------------------------------------------------
// Prep 1: f32 -> bf16 bulk convert
// ---------------------------------------------------------------------------
__global__ __launch_bounds__(256) void cvt_bf16_kernel(
    const float* __restrict__ src, __bf16* __restrict__ dst)
{
  int i4 = blockIdx.x * blockDim.x + threadIdx.x;
  v4f v = *(const v4f*)(src + (size_t)i4 * 4);
  __bf16* d = dst + (size_t)i4 * 4;
  d[0] = (__bf16)v.x; d[1] = (__bf16)v.y; d[2] = (__bf16)v.z; d[3] = (__bf16)v.w;
}

// ---------------------------------------------------------------------------
// Prep 2: W[k][n] f32 -> WT[n][k] bf16
// ---------------------------------------------------------------------------
__global__ __launch_bounds__(256) void transpose_w_kernel(
    const float* __restrict__ W, __bf16* __restrict__ WT, int N)
{
  int n = blockIdx.x * blockDim.x + threadIdx.x;
  int k = blockIdx.y;
  WT[(size_t)n * Cq + k] = (__bf16)W[(size_t)k * N + n];
}

// ---------------------------------------------------------------------------
// Kernel 1: QKV GEMM (bf16 x bf16 -> f32 acc), double-buffered async staging.
// Block: 256 thr = 8 waves; tile 64(M) x 128(N); wave tile 32x32.
// ---------------------------------------------------------------------------
#define LDA_S 40
#define LDB_S 40

__global__ __launch_bounds__(256) void qkv_gemm_kernel(
    const __bf16* __restrict__ A,     // Xbf [16384,1024]
    const __bf16* __restrict__ BT,    // WT  [3072,1024]
    const float* __restrict__ bias,
    __bf16* __restrict__ Qb, __bf16* __restrict__ Kb, __bf16* __restrict__ Vt)
{
  __shared__ __align__(16) __bf16 As[2][64 * LDA_S];
  __shared__ __align__(16) __bf16 Bs[2][128 * LDB_S];

  const int tid  = threadIdx.x;
  const int lane = tid & 31;
  const int wave = tid >> 5;
  const int l15  = lane & 15;
  const int hs   = (lane >> 4) & 1;
  const int blockM = blockIdx.y * 64;
  const int blockN = blockIdx.x * 128;
  const int wm = (wave & 1) * 32;
  const int wn = (wave >> 1) * 32;

  v8f c[2][2] = {};

  auto stage = [&](__bf16* Asb, __bf16* Bsb, int k0) {
    int m = tid >> 2, seg = (tid & 3) * 8;
    async_cp_b128(&Asb[m * LDA_S + seg], A + (size_t)(blockM + m) * Cq + k0 + seg);
    #pragma unroll
    for (int i = 0; i < 2; ++i) {
      int g = tid + i * 256;
      int n = g >> 2, sg = (g & 3) * 8;
      async_cp_b128(&Bsb[n * LDB_S + sg], BT + (size_t)(blockN + n) * Cq + k0 + sg);
    }
  };
  auto compute = [&](const __bf16* Asb, const __bf16* Bsb) {
    v16bf a[2], b[2];
    #pragma unroll
    for (int mt = 0; mt < 2; ++mt) {
      const __bf16* p = &Asb[(wm + mt * 16 + l15) * LDA_S + hs * 8];
      a[mt] = frag16(p, p + 16);
    }
    #pragma unroll
    for (int nt = 0; nt < 2; ++nt) {
      const __bf16* p = &Bsb[(wn + nt * 16 + l15) * LDB_S + hs * 8];
      b[nt] = frag16(p, p + 16);
    }
    #pragma unroll
    for (int mt = 0; mt < 2; ++mt)
      #pragma unroll
      for (int nt = 0; nt < 2; ++nt)
        c[mt][nt] = wmma_bf16(a[mt], b[nt], c[mt][nt]);
  };

  stage(As[0], Bs[0], 0);
  for (int k0 = 0; k0 < Cq; k0 += 64) {
    wait_async0();
    __syncthreads();
    if (k0 + 32 < Cq) stage(As[1], Bs[1], k0 + 32);
    compute(As[0], Bs[0]);
    wait_async0();
    __syncthreads();
    if (k0 + 64 < Cq) stage(As[0], Bs[0], k0 + 64);
    compute(As[1], Bs[1]);
  }

  // epilogue: routing key is wave-uniform -> force scalar branch
  const int bb = blockM >> 11;
  const int tb = (blockM & (Tq - 1)) + wm;
  #pragma unroll
  for (int nt = 0; nt < 2; ++nt) {
    int nbase = __builtin_amdgcn_readfirstlane(blockN + wn + nt * 16);
    int which = nbase >> 10;
    int hn    = (nbase & 1023) >> 6;
    int dbase = nbase & 63;
    float bv  = bias[nbase + l15];
    size_t head = (size_t)(bb * Hq + hn);
    if (which == 2) {
      __bf16* base = Vt + (head * HDq + dbase + l15) * Tq;   // [B,H,HD,T]
      #pragma unroll
      for (int mt = 0; mt < 2; ++mt)
        #pragma unroll
        for (int e = 0; e < 8; ++e)
          base[tb + mt * 16 + e + hs * 8] = (__bf16)(c[mt][nt][e] + bv);
    } else {
      __bf16* base = ((which == 1) ? Kb : Qb) + head * Tq * HDq + dbase + l15;
      #pragma unroll
      for (int mt = 0; mt < 2; ++mt)
        #pragma unroll
        for (int e = 0; e < 8; ++e)
          base[(size_t)(tb + mt * 16 + e + hs * 8) * HDq] = (__bf16)(c[mt][nt][e] + bv);
    }
  }
}

// ---------------------------------------------------------------------------
// Kernel 2: RoPE on Q and K in place.
// ---------------------------------------------------------------------------
__global__ __launch_bounds__(256) void rope_kernel(
    __bf16* __restrict__ Qb, __bf16* __restrict__ Kb,
    const float* __restrict__ freqs)
{
  int idx = blockIdx.x * blockDim.x + threadIdx.x;
  int i  = idx & 31;
  int t  = (idx >> 5) & (Tq - 1);
  int bh = idx >> 16;
  float f = freqs[t * 32 + i];
  float cs = cosf(f), sn = sinf(f);
  size_t base = ((size_t)bh * Tq + t) * HDq + i * 2;

  float q0 = (float)Qb[base], q1 = (float)Qb[base + 1];
  Qb[base]     = (__bf16)(q0 * cs - q1 * sn);
  Qb[base + 1] = (__bf16)(q0 * sn + q1 * cs);

  float k0 = (float)Kb[base], k1 = (float)Kb[base + 1];
  Kb[base]     = (__bf16)(k0 * cs - k1 * sn);
  Kb[base + 1] = (__bf16)(k0 * sn + k1 * cs);
}

// ---------------------------------------------------------------------------
// Kernel 3: flash attention, double-buffered async K/V staging.
// Grid: (T/64, B*H).  Block: 128 thr = 4 waves.
// ---------------------------------------------------------------------------
#define KS_S 72
#define VT_S 40
#define PS_S 40

__global__ __launch_bounds__(128) void attn_kernel(
    const __bf16* __restrict__ Qb, const __bf16* __restrict__ Kb,
    const __bf16* __restrict__ Vt, __bf16* __restrict__ Yb)
{
  __shared__ __align__(16) __bf16 Ks[2][32 * KS_S];
  __shared__ __align__(16) __bf16 Vs[2][64 * VT_S];
  __shared__ __align__(16) __bf16 Ps[4][16 * PS_S];

  const int tid  = threadIdx.x;
  const int lane = tid & 31;
  const int wave = tid >> 5;
  const int l15  = lane & 15;
  const int hs   = (lane >> 4) & 1;
  const int bh   = blockIdx.y;
  const int qb   = blockIdx.x;
  const int qwave = qb * 64 + wave * 16;
  const size_t head_off = (size_t)bh * Tq * HDq;

  const __bf16* qrow = Qb + head_off + (size_t)(qwave + l15) * HDq;
  v16bf aq[2];
  #pragma unroll
  for (int cc = 0; cc < 2; ++cc) {
    const __bf16* p = qrow + cc * 32 + hs * 8;
    aq[cc] = frag16(p, p + 16);
  }

  float m_i[8], l_i[8];
  #pragma unroll
  for (int e = 0; e < 8; ++e) { m_i[e] = -1e30f; l_i[e] = 0.0f; }
  v8f o[4] = {};

  auto stage = [&](__bf16* Ksb, __bf16* Vsb, int kv0) {
    #pragma unroll
    for (int i = 0; i < 2; ++i) {
      int g = tid + i * 128;
      int key = g >> 3, seg = (g & 7) * 8;
      async_cp_b128(&Ksb[key * KS_S + seg],
                    Kb + head_off + (size_t)(kv0 + key) * HDq + seg);
    }
    #pragma unroll
    for (int i = 0; i < 2; ++i) {
      int g = tid + i * 128;
      int d = g >> 2, seg = (g & 3) * 8;
      async_cp_b128(&Vsb[d * VT_S + seg],
                    Vt + head_off + (size_t)d * Tq + kv0 + seg);
    }
  };

  auto process = [&](const __bf16* Ksb, const __bf16* Vsb, int kv0) {
    // scores: two 16-key sub-tiles, K-reduction over HD in 2 chunks
    v8f s[2];
    #pragma unroll
    for (int j = 0; j < 2; ++j) {
      v8f acc = {};
      #pragma unroll
      for (int cc = 0; cc < 2; ++cc) {
        const __bf16* p = &Ksb[(j * 16 + l15) * KS_S + cc * 32 + hs * 8];
        v16bf bk = frag16(p, p + 16);
        acc = wmma_bf16(aq[cc], bk, acc);
      }
      s[j] = acc;
    }

    const float scale = 0.125f;
    float mchunk[8];
    #pragma unroll
    for (int e = 0; e < 8; ++e) {
      int qrow_g = qwave + e + hs * 8;
      #pragma unroll
      for (int j = 0; j < 2; ++j) {
        int col = kv0 + j * 16 + l15;
        float v = s[j][e] * scale;
        if (col > qrow_g) v = -1e30f;
        s[j][e] = v;
      }
      float mc = fmaxf(s[0][e], s[1][e]);
      #pragma unroll
      for (int off = 8; off >= 1; off >>= 1)
        mc = fmaxf(mc, __shfl_xor(mc, off, 16));
      mchunk[e] = mc;
    }

    float alpha[8];
    #pragma unroll
    for (int e = 0; e < 8; ++e) {
      float mnew = fmaxf(m_i[e], mchunk[e]);
      alpha[e] = __expf(m_i[e] - mnew);
      m_i[e] = mnew;
      float psum = 0.0f;
      #pragma unroll
      for (int j = 0; j < 2; ++j) {
        float p = __expf(s[j][e] - mnew);
        s[j][e] = p;
        psum += p;
      }
      #pragma unroll
      for (int off = 8; off >= 1; off >>= 1)
        psum += __shfl_xor(psum, off, 16);
      l_i[e] = l_i[e] * alpha[e] + psum;
    }
    #pragma unroll
    for (int jd = 0; jd < 4; ++jd)
      #pragma unroll
      for (int e = 0; e < 8; ++e)
        o[jd][e] *= alpha[e];

    // P: C-layout -> A-layout via per-wave LDS round trip
    #pragma unroll
    for (int e = 0; e < 8; ++e) {
      int r = e + hs * 8;
      Ps[wave][r * PS_S + l15]      = (__bf16)s[0][e];
      Ps[wave][r * PS_S + 16 + l15] = (__bf16)s[1][e];
    }
    asm volatile("s_wait_dscnt 0" ::: "memory");
    const __bf16* pp = &Ps[wave][l15 * PS_S + hs * 8];
    v16bf ap = frag16(pp, pp + 16);

    #pragma unroll
    for (int jd = 0; jd < 4; ++jd) {
      const __bf16* p = &Vsb[(jd * 16 + l15) * VT_S + hs * 8];
      v16bf bv = frag16(p, p + 16);
      o[jd] = wmma_bf16(ap, bv, o[jd]);
    }
  };

  const int kv_end = qb * 64 + 64;          // 2*(qb+1) chunks of 32: always even
  stage(Ks[0], Vs[0], 0);
  for (int kv0 = 0; kv0 < kv_end; kv0 += 64) {
    wait_async0();
    __syncthreads();
    if (kv0 + 32 < kv_end) stage(Ks[1], Vs[1], kv0 + 32);
    process(Ks[0], Vs[0], kv0);
    wait_async0();
    __syncthreads();
    if (kv0 + 64 < kv_end) stage(Ks[0], Vs[0], kv0 + 64);
    process(Ks[1], Vs[1], kv0 + 32);
  }

  int bb = bh >> 4, hh = bh & 15;
  float inv_l[8];
  #pragma unroll
  for (int e = 0; e < 8; ++e) inv_l[e] = 1.0f / l_i[e];
  #pragma unroll
  for (int jd = 0; jd < 4; ++jd) {
    #pragma unroll
    for (int e = 0; e < 8; ++e) {
      int t = qwave + e + hs * 8;
      Yb[((size_t)(bb * Tq + t)) * Cq + hh * HDq + jd * 16 + l15] =
          (__bf16)(o[jd][e] * inv_l[e]);
    }
  }
}

// ---------------------------------------------------------------------------
// Kernel 4: output projection, double-buffered async staging.
// ---------------------------------------------------------------------------
__global__ __launch_bounds__(256) void proj_gemm_kernel(
    const __bf16* __restrict__ A,     // Yb [16384,1024]
    const __bf16* __restrict__ BT,    // WTproj [1024,1024]
    const float* __restrict__ bias,
    float* __restrict__ Out)
{
  __shared__ __align__(16) __bf16 As[2][64 * LDA_S];
  __shared__ __align__(16) __bf16 Bs[2][128 * LDB_S];

  const int tid  = threadIdx.x;
  const int lane = tid & 31;
  const int wave = tid >> 5;
  const int l15  = lane & 15;
  const int hs   = (lane >> 4) & 1;
  const int blockM = blockIdx.y * 64;
  const int blockN = blockIdx.x * 128;
  const int wm = (wave & 1) * 32;
  const int wn = (wave >> 1) * 32;

  v8f c[2][2] = {};

  auto stage = [&](__bf16* Asb, __bf16* Bsb, int k0) {
    int m = tid >> 2, seg = (tid & 3) * 8;
    async_cp_b128(&Asb[m * LDA_S + seg], A + (size_t)(blockM + m) * Cq + k0 + seg);
    #pragma unroll
    for (int i = 0; i < 2; ++i) {
      int g = tid + i * 256;
      int n = g >> 2, sg = (g & 3) * 8;
      async_cp_b128(&Bsb[n * LDB_S + sg], BT + (size_t)(blockN + n) * Cq + k0 + sg);
    }
  };
  auto compute = [&](const __bf16* Asb, const __bf16* Bsb) {
    v16bf a[2], b[2];
    #pragma unroll
    for (int mt = 0; mt < 2; ++mt) {
      const __bf16* p = &Asb[(wm + mt * 16 + l15) * LDA_S + hs * 8];
      a[mt] = frag16(p, p + 16);
    }
    #pragma unroll
    for (int nt = 0; nt < 2; ++nt) {
      const __bf16* p = &Bsb[(wn + nt * 16 + l15) * LDB_S + hs * 8];
      b[nt] = frag16(p, p + 16);
    }
    #pragma unroll
    for (int mt = 0; mt < 2; ++mt)
      #pragma unroll
      for (int nt = 0; nt < 2; ++nt)
        c[mt][nt] = wmma_bf16(a[mt], b[nt], c[mt][nt]);
  };

  stage(As[0], Bs[0], 0);
  for (int k0 = 0; k0 < Cq; k0 += 64) {
    wait_async0();
    __syncthreads();
    if (k0 + 32 < Cq) stage(As[1], Bs[1], k0 + 32);
    compute(As[0], Bs[0]);
    wait_async0();
    __syncthreads();
    if (k0 + 64 < Cq) stage(As[0], Bs[0], k0 + 64);
    compute(As[1], Bs[1]);
  }

  #pragma unroll
  for (int nt = 0; nt < 2; ++nt) {
    float bv = bias[blockN + wn + nt * 16 + l15];
    float* base = Out + (size_t)(blockM + wm) * Cq + blockN + wn + nt * 16 + l15;
    #pragma unroll
    for (int mt = 0; mt < 2; ++mt)
      #pragma unroll
      for (int e = 0; e < 8; ++e)
        base[(size_t)(mt * 16 + e + hs * 8) * Cq] = c[mt][nt][e] + bv;
  }
}

// ---------------------------------------------------------------------------
extern "C" void kernel_launch(void* const* d_in, const int* in_sizes, int n_in,
                              void* d_out, int out_size, void* d_ws, size_t ws_size,
                              hipStream_t stream) {
  (void)in_sizes; (void)n_in; (void)out_size; (void)ws_size;
  const float* x      = (const float*)d_in[0];
  const float* freqs  = (const float*)d_in[1];
  const float* W_attn = (const float*)d_in[2];
  const float* b_attn = (const float*)d_in[3];
  const float* W_proj = (const float*)d_in[4];
  const float* b_proj = (const float*)d_in[5];
  float* out = (float*)d_out;

  char* ws = (char*)d_ws;
  const size_t SZ = (size_t)Bq * Hq * Tq * HDq * sizeof(__bf16);  // 32 MB
  __bf16* Qb  = (__bf16*)(ws);
  __bf16* Kb  = (__bf16*)(ws + SZ);
  __bf16* Vt  = (__bf16*)(ws + 2 * SZ);
  __bf16* Yb  = (__bf16*)(ws + 3 * SZ);
  __bf16* Xbf = (__bf16*)(ws + 4 * SZ);
  __bf16* WTa = (__bf16*)(ws + 5 * SZ);
  __bf16* WTp = (__bf16*)(ws + 5 * SZ + (size_t)3072 * 1024 * 2);

  cvt_bf16_kernel<<<(Bq * Tq * Cq) / 1024, 256, 0, stream>>>(x, Xbf);
  {
    dim3 g(3072 / 256, Cq);
    transpose_w_kernel<<<g, 256, 0, stream>>>(W_attn, WTa, 3072);
  }
  {
    dim3 g(Cq / 256, Cq);
    transpose_w_kernel<<<g, 256, 0, stream>>>(W_proj, WTp, Cq);
  }

  dim3 g1(3072 / 128, (Bq * Tq) / 64);
  qkv_gemm_kernel<<<g1, 256, 0, stream>>>(Xbf, WTa, b_attn, Qb, Kb, Vt);

  int rope_n = Bq * Hq * Tq * (HDq / 2);
  rope_kernel<<<rope_n / 256, 256, 0, stream>>>(Qb, Kb, freqs);

  dim3 g3(Tq / 64, Bq * Hq);
  attn_kernel<<<g3, 128, 0, stream>>>(Qb, Kb, Vt, Yb);

  dim3 g4(Cq / 128, (Bq * Tq) / 64);
  proj_gemm_kernel<<<g4, 256, 0, stream>>>(Yb, WTp, b_proj, out);
}
